// GRUModule_60146722013563
// MI455X (gfx1250) — matrix operand order repaired and checked by reading, not compiled, not run on hardware
//
#include <hip/hip_runtime.h>

typedef __attribute__((ext_vector_type(16))) __bf16 v16bf;
typedef __attribute__((ext_vector_type(8)))  float  v8f;

#define T_LEN 512
#define B_SZ  1024
#define D_IN  128
#define H_SZ  256
#define G3    768   // 3*H

// strides (padded to avoid LDS bank conflicts, keep 16B alignment for bf16 rows)
#define XBF_LD 136   // 128 + 8 bf16  (272B per row)
#define HBF_LD 264   // 256 + 8 bf16  (528B per row)
#define HFP_LD 257   // 256 + 1 f32
#define GB_LD  513   // 512 + 1 f32

__device__ __forceinline__ __bf16 f2bf(float f) {
  unsigned u = __float_as_uint(f);
  unsigned r = (u + 0x7FFFu + ((u >> 16) & 1u)) >> 16;   // round-to-nearest-even
  unsigned short s = (unsigned short)r;
  return __builtin_bit_cast(__bf16, s);
}

__device__ __forceinline__ float sigmoidf_(float x) {
  return 1.0f / (1.0f + __expf(-x));
}

__device__ __forceinline__ v8f wmma_bf16(v16bf a, v16bf b, v8f c) {
  // D = A(16x32 bf16) * B(32x16 bf16) + C(16x16 f32)
  return __builtin_amdgcn_wmma_f32_16x16x32_bf16(false, a, false, b, (short)0, c, false, false);
}

// A-matrix fragment (16x32 bf16) from LDS, row-major with stride ld.
__device__ __forceinline__ v16bf load_a_lds(const __bf16* base, int ld, int kbase, int lane) {
  int row = lane & 15, kh = (lane >> 4) & 1;
  const __bf16* p = base + row * ld + kbase + kh * 8;
  union { v16bf v; uint4 u[2]; } t;
  t.u[0] = *(const uint4*)(p);        // 8 bf16: k .. k+7
  t.u[1] = *(const uint4*)(p + 16);   // 8 bf16: k+16 .. k+23
  return t.v;
}

// B-matrix fragment (32x16 bf16): B[k][n] = W[colBase+n][kbase+k], W row-major ld.
__device__ __forceinline__ v16bf load_b_glb(const __bf16* W, int ld, int colBase, int kbase, int lane) {
  int n = lane & 15, kh = (lane >> 4) & 1;
  const __bf16* p = W + (size_t)(colBase + n) * ld + kbase + kh * 8;
  union { v16bf v; uint4 u[2]; } t;
  t.u[0] = *(const uint4*)(p);
  t.u[1] = *(const uint4*)(p + 16);
  return t.v;
}

__global__ void gru_cvt_weights(const float* __restrict__ wih,
                                const float* __restrict__ whh,
                                __bf16* __restrict__ out) {
  int i = blockIdx.x * blockDim.x + threadIdx.x;
  const int NIH = G3 * D_IN;           // 98304
  const int NALL = NIH + G3 * H_SZ;    // 294912
  if (i < NIH)       out[i] = f2bf(wih[i]);
  else if (i < NALL) out[i] = f2bf(whh[i - NIH]);
}

__global__ __launch_bounds__(256, 1)
void gru_scan_kernel(const float* __restrict__ x,
                     const float* __restrict__ rnn,
                     const float* __restrict__ masks,
                     const __bf16* __restrict__ wih_g,
                     const __bf16* __restrict__ whh_g,
                     const float* __restrict__ bih,
                     const float* __restrict__ bhh,
                     const float* __restrict__ lnw,
                     const float* __restrict__ lnb,
                     float* __restrict__ out) {
  __shared__ __bf16 s_xbf[16 * XBF_LD];   //  4352 B  x[t] tile as bf16
  __shared__ __bf16 s_hbf[16 * HBF_LD];   //  8448 B  h as bf16 (WMMA A)
  __shared__ float  s_hfp[16 * HFP_LD];   // 16448 B  h as f32 (master)
  __shared__ float  s_gbuf[16 * GB_LD];   // 32832 B  reused pre-activation buffer

  const int tid  = threadIdx.x;
  const int lane = tid & 31;
  const int wave = tid >> 5;          // 0..7
  const int nn   = lane & 15;
  const int kh   = (lane >> 4) & 1;
  const int rowbase = blockIdx.x * 16;

  const int crow = tid >> 4;          // combine-stage row 0..15
  const int cu0  = (tid & 15) * 16;   // combine-stage unit base
  const int crg  = rowbase + crow;    // global batch row for this thread's combine work

  // loop-invariant small params into registers
  float bv_i[6], bv_h[6];
  #pragma unroll
  for (int j = 0; j < 6; ++j) {
    int col = (wave * 6 + j) * 16 + nn;
    bv_i[j] = bih[col];
    bv_h[j] = bhh[col];
  }
  float lnw8[8], lnb8[8];
  {
    int c0 = lane * 8;
    #pragma unroll
    for (int i = 0; i < 8; ++i) { lnw8[i] = lnw[c0 + i]; lnb8[i] = lnb[c0 + i]; }
  }

  // init h from rnn_states[0]
  #pragma unroll
  for (int i = 0; i < 16; ++i)
    s_hfp[crow * HFP_LD + cu0 + i] = rnn[(size_t)crg * H_SZ + cu0 + i];

  for (int t = 0; t < T_LEN; ++t) {
    // Launder a zero *offset* (not the pointers): the weight pointers keep
    // their kernel-argument provenance, so clang still infers global address
    // space (global_load_b128, LOADcnt-only, saddr+voffset form), but the
    // loop-variant offset prevents hoisting all 72 weight fragments out of
    // the t-loop (which caused scratch spills).
    size_t zoff = 0;
    asm volatile("" : "+s"(zoff));
    const __bf16* wih = wih_g + zoff;
    const __bf16* whh = whh_g + zoff;

    __syncthreads();

    // ---- S0: apply mask to h, stage x[t] tile as bf16 ----
    {
      float m = masks[(size_t)t * B_SZ + crg];
      int d0 = (tid & 15) * 8;
      const float* xp = x + ((size_t)t * B_SZ + crg) * D_IN + d0;
      #pragma unroll
      for (int i = 0; i < 8; ++i)
        s_xbf[crow * XBF_LD + d0 + i] = f2bf(__builtin_nontemporal_load(xp + i));
      #pragma unroll
      for (int i = 0; i < 16; ++i) {
        int u = cu0 + i;
        float hv = s_hfp[crow * HFP_LD + u] * m;
        s_hfp[crow * HFP_LD + u] = hv;
        s_hbf[crow * HBF_LD + u] = f2bf(hv);
      }
    }
    __syncthreads();

    // ---- S1: WMMA  gi = x W_ih^T + b_ih ; gh = h W_hh^T + b_hh ----
    v8f gi[6], gh[6];
    {
      v16bf ax[4];
      #pragma unroll
      for (int kt = 0; kt < 4; ++kt)
        ax[kt] = load_a_lds(s_xbf, XBF_LD, kt * 32, lane);
      #pragma unroll
      for (int j = 0; j < 6; ++j) {
        int col = (wave * 6 + j) * 16;
        float bv = bv_i[j];
        v8f c = {bv, bv, bv, bv, bv, bv, bv, bv};
        #pragma unroll
        for (int kt = 0; kt < 4; ++kt)
          c = wmma_bf16(ax[kt], load_b_glb(wih, D_IN, col, kt * 32, lane), c);
        gi[j] = c;
      }
      v16bf ah[8];
      #pragma unroll
      for (int kt = 0; kt < 8; ++kt)
        ah[kt] = load_a_lds(s_hbf, HBF_LD, kt * 32, lane);
      #pragma unroll
      for (int j = 0; j < 6; ++j) {
        int col = (wave * 6 + j) * 16;
        float bv = bv_h[j];
        v8f c = {bv, bv, bv, bv, bv, bv, bv, bv};
        #pragma unroll
        for (int kt = 0; kt < 8; ++kt)
          c = wmma_bf16(ah[kt], load_b_glb(whh, H_SZ, col, kt * 32, lane), c);
        gh[j] = c;
      }
    }
    // write r/z pre-activation sums (gate cols 0..511) to gbuf
    #pragma unroll
    for (int j = 0; j < 6; ++j) {
      int tile = wave * 6 + j;
      if (tile < 32) {
        int col = tile * 16 + nn;
        #pragma unroll
        for (int c = 0; c < 8; ++c)
          s_gbuf[(c + 8 * kh) * GB_LD + col] = gi[j][c] + gh[j][c];
      }
    }
    __syncthreads();

    // ---- S2a: r,z into registers ----
    float rr[16], zz[16];
    #pragma unroll
    for (int i = 0; i < 16; ++i) {
      int u = cu0 + i;
      rr[i] = sigmoidf_(s_gbuf[crow * GB_LD + u]);
      zz[i] = sigmoidf_(s_gbuf[crow * GB_LD + 256 + u]);
    }
    __syncthreads();

    // ---- S2b: reuse gbuf for gi_n (cols 0..255) and gh_n (cols 256..511) ----
    #pragma unroll
    for (int j = 0; j < 6; ++j) {
      int tile = wave * 6 + j;
      if (tile >= 32) {
        int lc = (tile - 32) * 16 + nn;
        #pragma unroll
        for (int c = 0; c < 8; ++c) {
          s_gbuf[(c + 8 * kh) * GB_LD + lc]       = gi[j][c];
          s_gbuf[(c + 8 * kh) * GB_LD + 256 + lc] = gh[j][c];
        }
      }
    }
    __syncthreads();

    // ---- S2c: n = tanh(gi_n + r*gh_n); h = (1-z)*n + z*h ----
    #pragma unroll
    for (int i = 0; i < 16; ++i) {
      int u = cu0 + i;
      float gin = s_gbuf[crow * GB_LD + u];
      float ghn = s_gbuf[crow * GB_LD + 256 + u];
      float nv  = tanhf(gin + rr[i] * ghn);
      float hv  = (1.0f - zz[i]) * nv + zz[i] * s_hfp[crow * HFP_LD + u];
      s_hfp[crow * HFP_LD + u] = hv;
      s_hbf[crow * HBF_LD + u] = f2bf(hv);
    }
    __syncthreads();

    // ---- S3: fused LayerNorm + store y[t]; each wave handles 2 rows ----
    #pragma unroll
    for (int rp = 0; rp < 2; ++rp) {
      int row = wave * 2 + rp;
      int c0  = lane * 8;
      float v[8], sm = 0.f, sq = 0.f;
      #pragma unroll
      for (int i = 0; i < 8; ++i) {
        v[i] = s_hfp[row * HFP_LD + c0 + i];
        sm += v[i]; sq += v[i] * v[i];
      }
      for (int m2 = 16; m2 >= 1; m2 >>= 1) {
        sm += __shfl_xor(sm, m2, 32);
        sq += __shfl_xor(sq, m2, 32);
      }
      float mu  = sm * (1.0f / 256.0f);
      float var = sq * (1.0f / 256.0f) - mu * mu;
      float rs  = rsqrtf(var + 1e-5f);
      size_t base = ((size_t)t * B_SZ + rowbase + row) * H_SZ + c0;
      #pragma unroll
      for (int i = 0; i < 8; ++i)
        __builtin_nontemporal_store((v[i] - mu) * rs * lnw8[i] + lnb8[i], out + base + i);
    }
  }

  __syncthreads();
  // ---- h_final ----
  {
    size_t base = (size_t)T_LEN * B_SZ * H_SZ + (size_t)crg * H_SZ;
    #pragma unroll
    for (int i = 0; i < 16; ++i)
      out[base + cu0 + i] = s_hfp[crow * HFP_LD + cu0 + i];
  }
}

extern "C" void kernel_launch(void* const* d_in, const int* in_sizes, int n_in,
                              void* d_out, int out_size, void* d_ws, size_t ws_size,
                              hipStream_t stream) {
  const float* x     = (const float*)d_in[0];
  const float* rnn   = (const float*)d_in[1];
  const float* masks = (const float*)d_in[2];
  const float* Wih   = (const float*)d_in[3];
  const float* Whh   = (const float*)d_in[4];
  const float* bih   = (const float*)d_in[5];
  const float* bhh   = (const float*)d_in[6];
  const float* lnw   = (const float*)d_in[7];
  const float* lnb   = (const float*)d_in[8];
  float* out = (float*)d_out;

  __bf16* wbf    = (__bf16*)d_ws;
  __bf16* wih_bf = wbf;
  __bf16* whh_bf = wbf + G3 * D_IN;

  const int NALL = G3 * D_IN + G3 * H_SZ;   // 294912 elements
  gru_cvt_weights<<<dim3((NALL + 255) / 256), dim3(256), 0, stream>>>(Wih, Whh, wbf);
  gru_scan_kernel<<<dim3(B_SZ / 16), dim3(256), 0, stream>>>(
      x, rnn, masks, wih_bf, whh_bf, bih, bhh, lnw, lnb, out);
}
